// RSMLayer_47734266528347
// MI455X (gfx1250) — compile-verified, matrix operand order checked by reading, not run on hardware
//
#include <hip/hip_runtime.h>
#include <math.h>

// ---------------------------------------------------------------------------
// RSM layer for MI455X (gfx1250).
//   Batch-parallel Wa/Wd GEMMs hoisted out of the 16384-step scan onto
//   v_wmma_f32_16x16x32_f16 (f16 in / f32 acc): ping/pong software pipeline
//   (copy-free, x2 unrolled), b128 loads, 4x1 register tiling, unmasked
//   dead-column B (stores guarded). The sequential 1200x1200 matvec runs in
//   a persistent cooperative kernel with Wb cached in LDS (float4 ds reads)
//   and ONE grid sync per step (double-buffered sigma + redundant per-block
//   carry update).
// ---------------------------------------------------------------------------

typedef __attribute__((ext_vector_type(16))) _Float16 v16h;
typedef __attribute__((ext_vector_type(8)))  float    v8f;

#define Mg      200
#define Ng      6
#define KTOP    25
#define DIN     784
#define TOTALg  1200
#define BATCH   16384
#define GAMMAf  0.5f
#define EPSf    0.5f

#define SCAN_BLOCKS   150     // 150 * 8 rows = 1200 rows of Wb
#define ROWS_PER_BLK  8
#define SCAN_THREADS  256     // 8 wave32s

// ---------------------------- tile / pack helpers --------------------------
struct Tile { float4 q[4]; };          // 16 raw f32 K-values for one operand

__device__ inline float4 ldq(const float* p) { return *(const float4*)p; }

__device__ inline void ld_full(Tile& t, const float* p, int kl, int kh) {
  t.q[0] = ldq(p + kl); t.q[1] = ldq(p + kl + 4);
  t.q[2] = ldq(p + kh); t.q[3] = ldq(p + kh + 4);
}
__device__ inline void ld_half(Tile& t, const float* p, int kl) {   // hi K-half = 0
  t.q[0] = ldq(p + kl); t.q[1] = ldq(p + kl + 4);
  t.q[2] = make_float4(0.f, 0.f, 0.f, 0.f);
  t.q[3] = make_float4(0.f, 0.f, 0.f, 0.f);
}
__device__ inline void cvt8(v16h& d, int base, const float4 q0, const float4 q1) {
  d[base + 0] = (_Float16)q0.x; d[base + 1] = (_Float16)q0.y;
  d[base + 2] = (_Float16)q0.z; d[base + 3] = (_Float16)q0.w;
  d[base + 4] = (_Float16)q1.x; d[base + 5] = (_Float16)q1.y;
  d[base + 6] = (_Float16)q1.z; d[base + 7] = (_Float16)q1.w;
}
__device__ inline void cvt8m(v16h& d, int base, const float4 q0, const float4 q1,
                             bool m) {                       // cndmask, no branch
  d[base + 0] = (_Float16)(m ? q0.x : 0.f); d[base + 1] = (_Float16)(m ? q0.y : 0.f);
  d[base + 2] = (_Float16)(m ? q0.z : 0.f); d[base + 3] = (_Float16)(m ? q0.w : 0.f);
  d[base + 4] = (_Float16)(m ? q1.x : 0.f); d[base + 5] = (_Float16)(m ? q1.y : 0.f);
  d[base + 6] = (_Float16)(m ? q1.z : 0.f); d[base + 7] = (_Float16)(m ? q1.w : 0.f);
}
__device__ inline void zero8(v16h& d, int base) {
#pragma unroll
  for (int j = 0; j < 8; ++j) d[base + j] = (_Float16)0.f;
}
__device__ inline void cvt16(v16h& d, const Tile& t) {
  cvt8(d, 0, t.q[0], t.q[1]);
  cvt8(d, 8, t.q[2], t.q[3]);
}
__device__ inline void cvt16_tailm(v16h& d, const Tile& t, bool m) {
  cvt8m(d, 0, t.q[0], t.q[1], m);
  zero8(d, 8);
}

// Load one operand set (B + 4 A-tiles) for K-step k0 (full 32-wide).
__device__ inline void ld_set_full(Tile& tB, Tile& t0, Tile& t1, Tile& t2, Tile& t3,
                                   const float* wr, const float* r0, const float* r1,
                                   const float* r2, const float* r3,
                                   int k0, int khalf) {
  const int kl = k0 + khalf, kh = k0 + 16 + khalf;
  ld_full(tB, wr, kl, kh);
  ld_full(t0, r0, kl, kh);
  ld_full(t1, r1, kl, kh);
  ld_full(t2, r2, kl, kh);
  ld_full(t3, r3, kl, kh);
}
// Load one operand set with only the low K-half live (hi pre-zeroed).
__device__ inline void ld_set_half(Tile& tB, Tile& t0, Tile& t1, Tile& t2, Tile& t3,
                                   const float* wr, const float* r0, const float* r1,
                                   const float* r2, const float* r3, int kl) {
  ld_half(tB, wr, kl);
  ld_half(t0, r0, kl);
  ld_half(t1, r1, kl);
  ld_half(t2, r2, kl);
  ld_half(t3, r3, kl);
}
// Convert one buffered set and fire the 4 WMMAs.
__device__ inline void mma4(v8f& c0, v8f& c1, v8f& c2, v8f& c3,
                            const Tile& tB, const Tile& t0, const Tile& t1,
                            const Tile& t2, const Tile& t3) {
  v16h b, a0, a1, a2, a3;
  cvt16(b, tB);
  cvt16(a0, t0); cvt16(a1, t1); cvt16(a2, t2); cvt16(a3, t3);
  c0 = __builtin_amdgcn_wmma_f32_16x16x32_f16(false, a0, false, b, (short)0, c0, false, false);
  c1 = __builtin_amdgcn_wmma_f32_16x16x32_f16(false, a1, false, b, (short)0, c1, false, false);
  c2 = __builtin_amdgcn_wmma_f32_16x16x32_f16(false, a2, false, b, (short)0, c2, false, false);
  c3 = __builtin_amdgcn_wmma_f32_16x16x32_f16(false, a3, false, b, (short)0, c3, false, false);
}
// Same, with per-lane K mask on the low half (hi half compile-time zero).
__device__ inline void mma4_tailm(v8f& c0, v8f& c1, v8f& c2, v8f& c3,
                                  const Tile& tB, const Tile& t0, const Tile& t1,
                                  const Tile& t2, const Tile& t3, bool m) {
  v16h b, a0, a1, a2, a3;
  cvt16_tailm(b, tB, m);
  cvt16_tailm(a0, t0, m); cvt16_tailm(a1, t1, m);
  cvt16_tailm(a2, t2, m); cvt16_tailm(a3, t3, m);
  c0 = __builtin_amdgcn_wmma_f32_16x16x32_f16(false, a0, false, b, (short)0, c0, false, false);
  c1 = __builtin_amdgcn_wmma_f32_16x16x32_f16(false, a1, false, b, (short)0, c1, false, false);
  c2 = __builtin_amdgcn_wmma_f32_16x16x32_f16(false, a2, false, b, (short)0, c2, false, false);
  c3 = __builtin_amdgcn_wmma_f32_16x16x32_f16(false, a3, false, b, (short)0, c3, false, false);
}

// ------------------------------ grid sync ----------------------------------
__device__ inline void gsync(unsigned* cnt, unsigned* gen, unsigned nblk) {
  __syncthreads();
  __threadfence();                       // make sigma stores agent-visible
  if (threadIdx.x == 0) {
    unsigned g = __hip_atomic_load(gen, __ATOMIC_ACQUIRE, __HIP_MEMORY_SCOPE_AGENT);
    unsigned arrived = __hip_atomic_fetch_add(cnt, 1u, __ATOMIC_ACQ_REL,
                                              __HIP_MEMORY_SCOPE_AGENT);
    if (arrived == nblk - 1u) {
      __hip_atomic_store(cnt, 0u, __ATOMIC_RELAXED, __HIP_MEMORY_SCOPE_AGENT);
      __hip_atomic_store(gen, g + 1u, __ATOMIC_RELEASE, __HIP_MEMORY_SCOPE_AGENT);
    } else {
      while (__hip_atomic_load(gen, __ATOMIC_ACQUIRE, __HIP_MEMORY_SCOPE_AGENT) == g)
        __builtin_amdgcn_s_sleep(2);
    }
  }
  __syncthreads();
  __threadfence();                       // invalidate stale near caches for all lanes
}

// -------------------- Phase 1: Za = X @ Wa^T + ba  (WMMA) ------------------
// C is 16384x200. Each wave: 4 row-tiles (64 batch rows) x one 16-col M tile,
// sharing the B tile. K = 784 = 24 full 32-steps (12 ping/pong pairs) + tail.
// Dead B columns (mcol >= 200) unmasked: column n of D only sees column n of
// B, and those lanes' stores are guarded.
__global__ __launch_bounds__(256) void za_gemm(const float* __restrict__ X,
                                               const float* __restrict__ Wa,
                                               const float* __restrict__ ba,
                                               float* __restrict__ Za) {
  const int wave   = blockIdx.x * (blockDim.x >> 5) + (threadIdx.x >> 5);
  const int lane   = threadIdx.x & 31;
  const int tilesM = 13;                        // ceil(200/16)
  const int tg = wave / tilesM;                 // 64-row group (0..255)
  const int tm = wave % tilesM;                 // M tile
  const int lrow  = lane & 15;
  const int khalf = (lane >> 4) * 8;            // ISA A/B layout K half-split
  const int mcol  = tm * 16 + lrow;             // Wa row feeding B column
  const bool bval = (mcol < Mg);
  const float* __restrict__ wr  = Wa + (long)(bval ? mcol : (Mg - 1)) * DIN;
  const float* __restrict__ xr0 = X + (long)(tg * 64 +  0 + lrow) * DIN;
  const float* __restrict__ xr1 = X + (long)(tg * 64 + 16 + lrow) * DIN;
  const float* __restrict__ xr2 = X + (long)(tg * 64 + 32 + lrow) * DIN;
  const float* __restrict__ xr3 = X + (long)(tg * 64 + 48 + lrow) * DIN;

  Tile pB, p0, p1, p2, p3;                      // ping
  Tile qB, q0, q1, q2, q3;                      // pong
  ld_set_full(pB, p0, p1, p2, p3, wr, xr0, xr1, xr2, xr3, 0, khalf);

  v8f c0 = {}, c1 = {}, c2 = {}, c3 = {};
  for (int it = 0; it < 24; it += 2) {          // 12 copy-free ping/pong pairs
    // even step: prefetch pong(k=it+1), compute ping
    ld_set_full(qB, q0, q1, q2, q3, wr, xr0, xr1, xr2, xr3, (it + 1) * 32, khalf);
    mma4(c0, c1, c2, c3, pB, p0, p1, p2, p3);
    // odd step: prefetch ping(k=it+2 or K-tail), compute pong
    if (it + 2 < 24)
      ld_set_full(pB, p0, p1, p2, p3, wr, xr0, xr1, xr2, xr3, (it + 2) * 32, khalf);
    else
      ld_set_half(pB, p0, p1, p2, p3, wr, xr0, xr1, xr2, xr3, 768 + khalf);
    mma4(c0, c1, c2, c3, qB, q0, q1, q2, q3);
  }
  mma4(c0, c1, c2, c3, pB, p0, p1, p2, p3);     // K tail 768..783 (hi half zero)

  if (bval) {
    const int rb  = tg * 64 + ((lane >> 4) * 8);   // C/D layout: VGPR v -> row v (+8)
    const float bav = ba[mcol];
#pragma unroll
    for (int v = 0; v < 8; ++v) {
      Za[(long)(rb +  0 + v) * Mg + mcol] = c0[v] + bav;
      Za[(long)(rb + 16 + v) * Mg + mcol] = c1[v] + bav;
      Za[(long)(rb + 32 + v) * Mg + mcol] = c2[v] + bav;
      Za[(long)(rb + 48 + v) * Mg + mcol] = c3[v] + bav;
    }
  }
}

// ---------------- Phase 3: preds = Ymax @ Wd^T + bd  (WMMA) ----------------
// C is 16384x784 (49 exact col tiles). K = 200 = 6 full 32-steps (3 pairs) +
// 8 tail: tail raws load indices 192..199 (valid for every lane), masked by
// the per-half predicate khalf==0 (K contributions cross lanes, mask needed).
__global__ __launch_bounds__(256) void pred_gemm(const float* __restrict__ Y,
                                                 const float* __restrict__ Wd,
                                                 const float* __restrict__ bd,
                                                 float* __restrict__ preds) {
  const int wave   = blockIdx.x * (blockDim.x >> 5) + (threadIdx.x >> 5);
  const int lane   = threadIdx.x & 31;
  const int tilesD = DIN / 16;                  // 49
  const int tg = wave / tilesD;                 // 64-row group (0..255)
  const int td = wave % tilesD;
  const int lrow  = lane & 15;
  const int khalf = (lane >> 4) * 8;
  const int dcol  = td * 16 + lrow;             // < 784 always
  const float* __restrict__ wr  = Wd + (long)dcol * Mg;   // Wd[d][m], contiguous in m
  const float* __restrict__ yr0 = Y + (long)(tg * 64 +  0 + lrow) * Mg;
  const float* __restrict__ yr1 = Y + (long)(tg * 64 + 16 + lrow) * Mg;
  const float* __restrict__ yr2 = Y + (long)(tg * 64 + 32 + lrow) * Mg;
  const float* __restrict__ yr3 = Y + (long)(tg * 64 + 48 + lrow) * Mg;

  Tile pB, p0, p1, p2, p3;                      // ping
  Tile qB, q0, q1, q2, q3;                      // pong
  ld_set_full(pB, p0, p1, p2, p3, wr, yr0, yr1, yr2, yr3, 0, khalf);

  v8f c0 = {}, c1 = {}, c2 = {}, c3 = {};
  const bool lo = (khalf == 0);
  for (int it = 0; it < 6; it += 2) {           // 3 copy-free ping/pong pairs
    ld_set_full(qB, q0, q1, q2, q3, wr, yr0, yr1, yr2, yr3, (it + 1) * 32, khalf);
    mma4(c0, c1, c2, c3, pB, p0, p1, p2, p3);
    if (it + 2 < 6)
      ld_set_full(pB, p0, p1, p2, p3, wr, yr0, yr1, yr2, yr3, (it + 2) * 32, khalf);
    else
      ld_set_half(pB, p0, p1, p2, p3, wr, yr0, yr1, yr2, yr3, 192);
    mma4(c0, c1, c2, c3, qB, q0, q1, q2, q3);
  }
  mma4_tailm(c0, c1, c2, c3, pB, p0, p1, p2, p3, lo);   // K 192..199

  const int rb = tg * 64 + ((lane >> 4) * 8);
  const float bdv = bd[dcol];
#pragma unroll
  for (int v = 0; v < 8; ++v) {                 // write-once output: NT stores
    __builtin_nontemporal_store(c0[v] + bdv, &preds[(long)(rb +  0 + v) * DIN + dcol]);
    __builtin_nontemporal_store(c1[v] + bdv, &preds[(long)(rb + 16 + v) * DIN + dcol]);
    __builtin_nontemporal_store(c2[v] + bdv, &preds[(long)(rb + 32 + v) * DIN + dcol]);
    __builtin_nontemporal_store(c3[v] + bdv, &preds[(long)(rb + 48 + v) * DIN + dcol]);
  }
}

// -------------------- Phase 2: persistent sequential scan ------------------
__global__ __launch_bounds__(SCAN_THREADS) void scan_kernel(
    const float* __restrict__ Wb, const float* __restrict__ bb,
    const float* __restrict__ Za, float* __restrict__ Ymax,
    float* __restrict__ sigbuf,              // 2 * TOTALg (double buffer)
    float* __restrict__ outTail,             // x_b | phi | psi (block 0 writes)
    unsigned* sync_cnt, unsigned* sync_gen) {
  __shared__ __align__(16) float wlds[ROWS_PER_BLK * TOTALg];   // 38.4 KB Wb rows
  __shared__ __align__(16) float xb[TOTALg];                    // private carry
  __shared__ float phi[TOTALg];
  __shared__ float psi[TOTALg];
  __shared__ float sig[TOTALg];
  __shared__ float lamv[Mg];
  __shared__ float sigAt[Mg];
  __shared__ int   argn[Mg];
  __shared__ int   sel[Mg];
  __shared__ float redv[8];
  __shared__ int   redi[8];
  __shared__ float scal[2];                       // [0]=sigma min, [1]=alpha

  const int tid  = threadIdx.x;
  const int bid  = blockIdx.x;
  const int lane = tid & 31;
  const int wv   = tid >> 5;
  const int row0 = bid * ROWS_PER_BLK;

  // Stage our Wb slice into LDS once (streams once from L2-resident Wb).
  for (int i = tid; i < ROWS_PER_BLK * TOTALg; i += SCAN_THREADS)
    wlds[i] = Wb[(long)row0 * TOTALg + i];
  for (int i = tid; i < TOTALg; i += SCAN_THREADS) { xb[i] = 0.f; phi[i] = 0.f; psi[i] = 0.f; }
  __syncthreads();

  for (int t = 0; t < BATCH; ++t) {
    float* sgout = sigbuf + (t & 1) * TOTALg;

    // ---- Phase A: our 8 rows of sigma = Wb@xb + bb + broadcast(Za[t]) ----
    for (int rl = wv; rl < ROWS_PER_BLK; rl += (SCAN_THREADS / 32)) {
      const int r = row0 + rl;
      const float4* wr4 = (const float4*)(&wlds[rl * TOTALg]);   // 4.8KB rows: 16B aligned
      const float4* xb4 = (const float4*)xb;
      float p = 0.f;
      for (int c = lane; c < TOTALg / 4; c += 32) {              // ds_load_b128 x2
        float4 w = wr4[c], x = xb4[c];
        p += w.x * x.x + w.y * x.y + w.z * x.z + w.w * x.w;
      }
      for (int off = 16; off > 0; off >>= 1) p += __shfl_xor(p, off, 32);
      if (lane == 0) sgout[r] = p + bb[r] + Za[(long)t * Mg + (r / Ng)];
    }
    gsync(sync_cnt, sync_gen, SCAN_BLOCKS);   // the single sync per step

    // ---- Phase B (redundant per block; keeps carry local, avoids 2nd sync)
    for (int i = tid; i < TOTALg; i += SCAN_THREADS) sig[i] = sgout[i];
    __syncthreads();

    // sigma.min()
    float lmin = 3.4e38f;
    for (int i = tid; i < TOTALg; i += SCAN_THREADS) lmin = fminf(lmin, sig[i]);
    for (int off = 16; off > 0; off >>= 1) lmin = fminf(lmin, __shfl_xor(lmin, off, 32));
    if (lane == 0) redv[wv] = lmin;
    __syncthreads();
    if (tid == 0) { float mm = redv[0]; for (int i = 1; i < 8; ++i) mm = fminf(mm, redv[i]); scal[0] = mm; }
    __syncthreads();
    const float smin = scal[0];

    // per-group pi, row argmax -> lambda
    if (tid < Mg) {
      float best = -3.4e38f; int bestn = 0;
#pragma unroll
      for (int n = 0; n < Ng; ++n) {
        float s = sig[tid * Ng + n];
        float p = (1.f - phi[tid * Ng + n]) * (s - smin + 1.f);
        if (p > best) { best = p; bestn = n; }
      }
      lamv[tid] = best; argn[tid] = bestn; sigAt[tid] = sig[tid * Ng + bestn]; sel[tid] = 0;
    }
    __syncthreads();

    // top-25 groups: 25 block-wide argmax passes over 200 lambdas
    for (int kk = 0; kk < KTOP; ++kk) {
      float v  = (tid < Mg && !sel[tid]) ? lamv[tid] : -3.4e38f;
      int   ix = tid;
      for (int off = 16; off > 0; off >>= 1) {
        float ov = __shfl_xor(v, off, 32);
        int   oi = __shfl_xor(ix, off, 32);
        if (ov > v) { v = ov; ix = oi; }
      }
      if (lane == 0) { redv[wv] = v; redi[wv] = ix; }
      __syncthreads();
      if (tid == 0) {
        float bv = redv[0]; int bi = redi[0];
        for (int i = 1; i < 8; ++i) if (redv[i] > bv) { bv = redv[i]; bi = redi[i]; }
        sel[bi] = 1;
      }
      __syncthreads();
    }

    // y = tanh(masked sigma); psi/phi decay-max; block0 writes Ymax[t]
    if (tid < Mg) {
      const float yv = sel[tid] ? tanhf(sigAt[tid]) : 0.f;
      const int an = argn[tid];
#pragma unroll
      for (int n = 0; n < Ng; ++n) {
        float ymn = (sel[tid] && n == an) ? yv : 0.f;
        psi[tid * Ng + n] = fmaxf(EPSf   * psi[tid * Ng + n], ymn);
        phi[tid * Ng + n] = fmaxf(GAMMAf * phi[tid * Ng + n], ymn);
      }
      if (bid == 0) Ymax[(long)t * Mg + tid] = sel[tid] ? fmaxf(yv, 0.f) : 0.f;
    }
    __syncthreads();

    // alpha = sum(psi) (0 -> 1); xb = psi / alpha
    float ls = 0.f;
    for (int i = tid; i < TOTALg; i += SCAN_THREADS) ls += psi[i];
    for (int off = 16; off > 0; off >>= 1) ls += __shfl_xor(ls, off, 32);
    if (lane == 0) redv[wv] = ls;
    __syncthreads();
    if (tid == 0) { float a = 0.f; for (int i = 0; i < 8; ++i) a += redv[i]; scal[1] = (a == 0.f) ? 1.f : a; }
    __syncthreads();
    const float inva = 1.f / scal[1];
    for (int i = tid; i < TOTALg; i += SCAN_THREADS) xb[i] = psi[i] * inva;
    __syncthreads();
  }

  if (bid == 0) {
    for (int i = tid; i < TOTALg; i += SCAN_THREADS) {
      outTail[i]              = xb[i];
      outTail[TOTALg + i]     = phi[i];
      outTail[2 * TOTALg + i] = psi[i];
    }
  }
}

__global__ void init_sync(unsigned* sync_cnt, unsigned* sync_gen) {
  if (threadIdx.x == 0) { sync_cnt[0] = 0u; sync_gen[0] = 0u; }
}

// ------------------------------- launcher ----------------------------------
extern "C" void kernel_launch(void* const* d_in, const int* in_sizes, int n_in,
                              void* d_out, int out_size, void* d_ws, size_t ws_size,
                              hipStream_t stream) {
  const float* X  = (const float*)d_in[0];   // (16384, 784)
  const float* Wa = (const float*)d_in[1];   // (200, 784)
  const float* ba = (const float*)d_in[2];   // (200,)
  const float* Wb = (const float*)d_in[3];   // (1200, 1200)
  const float* bb = (const float*)d_in[4];   // (1200,)
  const float* Wd = (const float*)d_in[5];   // (784, 200)
  const float* bd = (const float*)d_in[6];   // (784,)
  float* out = (float*)d_out;                // preds | x_b | phi | psi

  float* ws      = (float*)d_ws;
  float* Za      = ws;                              // 16384*200
  float* Ymax    = Za + (long)BATCH * Mg;           // 16384*200
  float* sigbuf  = Ymax + (long)BATCH * Mg;         // 2*1200
  unsigned* sync = (unsigned*)(sigbuf + 2 * TOTALg);

  init_sync<<<1, 32, 0, stream>>>(sync, sync + 1);

  // Phase 1: hoisted Wa GEMM. 256 row-groups x 13 M-tiles, 8 waves/block.
  za_gemm<<<(256 * 13) / 8, 256, 0, stream>>>(X, Wa, ba, Za);

  // Phase 2: sequential scan, 150 persistent blocks, 1 grid sync/step.
  scan_kernel<<<SCAN_BLOCKS, SCAN_THREADS, 0, stream>>>(
      Wb, bb, Za, Ymax, sigbuf, out + (long)BATCH * DIN, sync, sync + 1);

  // Phase 3: hoisted Wd GEMM. 256 row-groups x 49 D-tiles, 8 waves/block.
  pred_gemm<<<(256 * 49) / 8, 256, 0, stream>>>(Ymax, Wd, bd, out);
}
